// SparseMoE_81905026334892
// MI455X (gfx1250) — matrix operand order repaired and checked by reading, not compiled, not run on hardware
//
#include <hip/hip_runtime.h>
#include <hip/hip_bf16.h>

// ---------------- problem constants ----------------
#define DM    1024          // d_model
#define HID   2048          // hidden = 2*d_model
#define NE    8             // experts
#define NTOK  8192          // B*S tokens
#define CT1   128           // HID/16 column tiles of W1
#define KT1   32            // DM/32  k tiles of W1
#define CT2   64            // DM/16  column tiles of W2
#define KT2   64            // HID/32 k tiles of W2
#define MTILE 32            // tokens per block (2 x 16-row WMMA subtiles)
#define MAXT  256           // max 32-token tiles per expert (NTOK/32)
#define XS_STRIDE 1032      // 1024 + 8 bf16 pad (bank spread)
#define HS_STRIDE 264       // 256 + 8 bf16 pad

// workspace byte offsets (total ~84.5 MB)
#define OFF_COUNTS 0u
#define OFF_TOK    256u
#define OFF_LW     (256u + 262144u)
#define OFF_XB     524544u
#define OFF_W1P    17301760u
#define OFF_W2P    50856192u

typedef __attribute__((ext_vector_type(16))) __bf16 v16bf;
typedef __attribute__((ext_vector_type(8)))  float  v8f;

union FragA { v16bf v; unsigned int u[8]; };
union FragB { v16bf v; uint4 q[2]; };
union FragC { v8f   v; float f[8]; };

__device__ __forceinline__ unsigned short f32_to_bf16_rne(float f) {
  union { float f; unsigned int u; } x; x.f = f;
  unsigned int u = x.u;
  if ((u & 0x7fffffffu) > 0x7f800000u) return (unsigned short)((u >> 16) | 0x0040u); // NaN
  u += 0x7fffu + ((u >> 16) & 1u);
  return (unsigned short)(u >> 16);
}

// ---- gfx1250 async mem->LDS staging (guarded; falls back to plain copy) ----
#if defined(__has_builtin)
#if __has_builtin(__builtin_amdgcn_global_load_async_to_lds_b128) && \
    __has_builtin(__builtin_amdgcn_s_wait_asynccnt)
#define USE_ASYNC_LDS 1
#endif
#endif

typedef int v4i_ __attribute__((vector_size(16)));
typedef __attribute__((address_space(1))) v4i_* async_gptr;   // non-const per diagnostic
typedef __attribute__((address_space(3))) v4i_* async_lptr;

__device__ __forceinline__ void lds_copy16(const void* g, void* l) {
#ifdef USE_ASYNC_LDS
  __builtin_amdgcn_global_load_async_to_lds_b128(
      (async_gptr)(unsigned long long)(uintptr_t)g,
      (async_lptr)(unsigned int)(uintptr_t)l,
      0, 0);
#else
  *(uint4*)l = *(const uint4*)g;
#endif
}
__device__ __forceinline__ void lds_copy_wait() {
#ifdef USE_ASYNC_LDS
  __builtin_amdgcn_s_wait_asynccnt(0);
#endif
}

// ---------------- kernel 1: zero output + routing counters ----------------
__global__ void k_zero(float4* __restrict__ out4, unsigned int* __restrict__ counts) {
  int i = blockIdx.x * 256 + threadIdx.x;              // 2,097,152 float4 = 8M floats
  out4[i] = make_float4(0.f, 0.f, 0.f, 0.f);
  if (blockIdx.x == 0 && threadIdx.x < NE) counts[threadIdx.x] = 0u;
}

// ---------------- kernel 2: gating + routing (top-2, renormalized) -------
__global__ __launch_bounds__(256) void k_gate(const float* __restrict__ x,
                                              const float* __restrict__ Wg,
                                              const float* __restrict__ bg,
                                              int* __restrict__ ltok,
                                              float* __restrict__ lw,
                                              unsigned int* __restrict__ counts) {
  __shared__ float wgs[DM * NE];                        // 32 KB
  for (int i = threadIdx.x; i < DM * NE; i += 256) wgs[i] = Wg[i];
  __syncthreads();

  const int wave = threadIdx.x >> 5;
  const int lane = threadIdx.x & 31;
  const int token = blockIdx.x * 8 + wave;
  const float* xr = x + (size_t)token * DM;

  float acc[NE];
#pragma unroll
  for (int e = 0; e < NE; ++e) acc[e] = 0.f;

  for (int j = lane; j < DM; j += 32) {
    const float xv = xr[j];
#pragma unroll
    for (int e = 0; e < NE; ++e) acc[e] += xv * wgs[j * NE + e];
  }
#pragma unroll
  for (int e = 0; e < NE; ++e) {
#pragma unroll
    for (int off = 16; off > 0; off >>= 1) acc[e] += __shfl_xor(acc[e], off, 32);
  }
  if (lane == 0) {
    float l[NE];
#pragma unroll
    for (int e = 0; e < NE; ++e) l[e] = acc[e] + bg[e];
    int i0 = 0;
#pragma unroll
    for (int e = 1; e < NE; ++e) if (l[e] > l[i0]) i0 = e;
    int i1 = (i0 == 0) ? 1 : 0;
#pragma unroll
    for (int e = 0; e < NE; ++e) if (e != i0 && l[e] > l[i1]) i1 = e;
    const float d  = __expf(l[i1] - l[i0]);
    const float w0 = 1.f / (1.f + d);
    const float w1 = d / (1.f + d);
    unsigned p0 = atomicAdd(&counts[i0], 1u);
    ltok[i0 * NTOK + p0] = token; lw[i0 * NTOK + p0] = w0;
    unsigned p1 = atomicAdd(&counts[i1], 1u);
    ltok[i1 * NTOK + p1] = token; lw[i1 * NTOK + p1] = w1;
  }
}

// ---------------- kernel 3: x fp32 -> bf16 row-major ---------------------
__global__ void k_cvt_x(const float4* __restrict__ x4, ushort4* __restrict__ xb4) {
  int i = blockIdx.x * 256 + threadIdx.x;               // 2,097,152
  float4 v = x4[i];
  ushort4 o;
  o.x = f32_to_bf16_rne(v.x); o.y = f32_to_bf16_rne(v.y);
  o.z = f32_to_bf16_rne(v.z); o.w = f32_to_bf16_rne(v.w);
  xb4[i] = o;
}

// ---------------- kernels 4/5: pack W1/W2 into WMMA B-fragment layout ----
// B (32x16 bf16): column n = lane&15; half = lane>>4; element j holds
// k = kt*32 + half*16 + j  (this code's contiguous order: VGPR v = rows k=2v,2v+1 per half; A matches).
__global__ void k_pack_w1(const float* __restrict__ W1, unsigned short* __restrict__ W1p) {
  unsigned flat = blockIdx.x * 256u + threadIdx.x;      // 8*128*32*32 = 1,048,576
  unsigned lane = flat & 31u; unsigned r = flat >> 5;
  unsigned kt = r & (KT1 - 1); r >>= 5;
  unsigned ct = r & (CT1 - 1); unsigned e = r >> 7;
  unsigned hi = lane >> 4;
  unsigned n  = ct * 16 + (lane & 15);
  const float* src = W1 + (size_t)e * DM * HID + n;
  union { unsigned short s[16]; uint4 q[2]; } tmp;
#pragma unroll
  for (int j = 0; j < 16; ++j) {
    unsigned k = kt * 32 + hi * 16 + (unsigned)j;
    tmp.s[j] = f32_to_bf16_rne(src[(size_t)k * HID]);
  }
  uint4* dst = (uint4*)(W1p + (size_t)flat * 16);
  dst[0] = tmp.q[0]; dst[1] = tmp.q[1];
}

__global__ void k_pack_w2(const float* __restrict__ W2, unsigned short* __restrict__ W2p) {
  unsigned flat = blockIdx.x * 256u + threadIdx.x;      // 8*64*64*32 = 1,048,576
  unsigned lane = flat & 31u; unsigned r = flat >> 5;
  unsigned kt = r & (KT2 - 1); r >>= 6;
  unsigned ct = r & (CT2 - 1); unsigned e = r >> 6;
  unsigned hi = lane >> 4;
  unsigned n  = ct * 16 + (lane & 15);
  const float* src = W2 + (size_t)e * HID * DM + n;
  union { unsigned short s[16]; uint4 q[2]; } tmp;
#pragma unroll
  for (int j = 0; j < 16; ++j) {
    unsigned k = kt * 32 + hi * 16 + (unsigned)j;
    tmp.s[j] = f32_to_bf16_rne(src[(size_t)k * DM]);
  }
  uint4* dst = (uint4*)(W2p + (size_t)flat * 16);
  dst[0] = tmp.q[0]; dst[1] = tmp.q[1];
}

// ---------------- kernel 6: routed fused MoE MLP (WMMA, M=32) ------------
// block = (expert e, 32-token tile); 8 waves; two 16-row subtiles share every
// B fragment (halves L2 weight traffic vs M=16). Fused GEMM1->relu->GEMM2.
__global__ __launch_bounds__(256) void k_moe(const unsigned short* __restrict__ xb,
                                             const unsigned int* __restrict__ w1p,
                                             const unsigned int* __restrict__ w2p,
                                             const float* __restrict__ b1,
                                             const float* __restrict__ b2,
                                             const int* __restrict__ ltok,
                                             const float* __restrict__ lw,
                                             const unsigned int* __restrict__ counts,
                                             float* __restrict__ out) {
  const int e = blockIdx.x >> 8;
  const int t = blockIdx.x & (MAXT - 1);
  const int cnt = (int)counts[e];
  if (t * MTILE >= cnt) return;

  extern __shared__ char smem[];
  unsigned short* xs  = (unsigned short*)smem;                       // 32*1032*2 = 66,048 B
  unsigned short* hs  = (unsigned short*)(smem + 66048);             // 32*264*2  = 16,896 B
  int*   toks = (int*)(smem + 82944);                                // 128 B
  float* tw   = (float*)(smem + 83072);                              // 128 B

  const int tid  = threadIdx.x;
  const int wave = tid >> 5;
  const int lane = tid & 31;
  const int hi   = lane >> 4;
  const int nlo  = lane & 15;

  if (tid < MTILE) {
    const int idx = t * MTILE + tid;
    if (idx < cnt) { toks[tid] = ltok[e * NTOK + idx]; tw[tid] = lw[e * NTOK + idx]; }
    else           { toks[tid] = 0;                    tw[tid] = 0.f; }
  }
  __syncthreads();

  // stage gathered 32x1024 bf16 x-tile into LDS (async mem->LDS when available)
  {
    const int r = tid >> 3, s = tid & 7;                 // 32 rows x 8 segs of 256 B
    const char* src = (const char*)(xb + (size_t)toks[r] * DM) + s * 256;
    char* dst = (char*)(xs + r * XS_STRIDE) + s * 256;
#pragma unroll
    for (int i = 0; i < 16; ++i) lds_copy16(src + i * 16, dst + i * 16);
  }
  lds_copy_wait();
  __syncthreads();

  FragC yacc[8][2];                                      // 8 y-tiles x 2 M-subtiles
#pragma unroll
  for (int i = 0; i < 8; ++i)
#pragma unroll
    for (int m = 0; m < 2; ++m)
#pragma unroll
      for (int j = 0; j < 8; ++j) yacc[i][m].f[j] = 0.f;

  const unsigned short* xrow0 = xs + nlo * XS_STRIDE;            // subtile 0 row
  const unsigned short* xrow1 = xs + (16 + nlo) * XS_STRIDE;     // subtile 1 row
  const unsigned short* hrow0 = hs + nlo * HS_STRIDE;
  const unsigned short* hrow1 = hs + (16 + nlo) * HS_STRIDE;

  for (int c = 0; c < 8; ++c) {                          // 256-col chunk of h
    // ---- phase A: h[:, chunk] = relu(x @ W1 + b1); wave owns 2 col tiles
    FragC hacc[2][2];                                    // [ctl][msub]
#pragma unroll
    for (int a = 0; a < 2; ++a)
#pragma unroll
      for (int m = 0; m < 2; ++m)
#pragma unroll
        for (int j = 0; j < 8; ++j) hacc[a][m].f[j] = 0.f;

    const int ctb = c * 16 + wave * 2;
    const unsigned int* bp0 = w1p + ((((size_t)e * CT1 + ctb)     * KT1) * 32 + lane) * 8;
    const unsigned int* bp1 = w1p + ((((size_t)e * CT1 + ctb + 1) * KT1) * 32 + lane) * 8;
    for (int kt = 0; kt < KT1; ++kt) {
      FragA a0, a1;
      const int k0 = kt * 32;
#pragma unroll
      for (int v = 0; v < 8; ++v) {
        const int k = k0 + hi * 16 + v * 2;
        a0.u[v] = *(const unsigned int*)(xrow0 + k);
        a1.u[v] = *(const unsigned int*)(xrow1 + k);
      }
      FragB b0, b1f;
      b0.q[0]  = ((const uint4*)bp0)[0]; b0.q[1]  = ((const uint4*)bp0)[1];
      b1f.q[0] = ((const uint4*)bp1)[0]; b1f.q[1] = ((const uint4*)bp1)[1];
      hacc[0][0].v = __builtin_amdgcn_wmma_f32_16x16x32_bf16(false, a0.v, false, b0.v,  (short)0, hacc[0][0].v, false, false);
      hacc[0][1].v = __builtin_amdgcn_wmma_f32_16x16x32_bf16(false, a1.v, false, b0.v,  (short)0, hacc[0][1].v, false, false);
      hacc[1][0].v = __builtin_amdgcn_wmma_f32_16x16x32_bf16(false, a0.v, false, b1f.v, (short)0, hacc[1][0].v, false, false);
      hacc[1][1].v = __builtin_amdgcn_wmma_f32_16x16x32_bf16(false, a1.v, false, b1f.v, (short)0, hacc[1][1].v, false, false);
      bp0 += 32 * 8; bp1 += 32 * 8;
    }
#pragma unroll
    for (int ctl = 0; ctl < 2; ++ctl) {
      const int ct   = ctb + ctl;
      const int col  = ct * 16 + nlo;
      const float bs = b1[e * HID + col];
      const int lcol = (wave * 2 + ctl) * 16 + nlo;
#pragma unroll
      for (int m = 0; m < 2; ++m)
#pragma unroll
        for (int r = 0; r < 8; ++r) {
          float hv = hacc[ctl][m].f[r] + bs;
          hv = hv > 0.f ? hv : 0.f;
          hs[(m * 16 + r + hi * 8) * HS_STRIDE + lcol] = f32_to_bf16_rne(hv);
        }
    }
    __syncthreads();
    // ---- phase B: y += h_chunk @ W2[chunk,:]; wave owns 8 y col tiles
    for (int kt = 0; kt < 8; ++kt) {
      FragA a0, a1;
      const int k0 = kt * 32;
#pragma unroll
      for (int v = 0; v < 8; ++v) {
        const int k = k0 + hi * 16 + v * 2;
        a0.u[v] = *(const unsigned int*)(hrow0 + k);
        a1.u[v] = *(const unsigned int*)(hrow1 + k);
      }
      const int ktg = c * 8 + kt;
#pragma unroll
      for (int t2 = 0; t2 < 8; ++t2) {
        const int ct2 = wave * 8 + t2;
        const unsigned int* bp =
            w2p + ((((size_t)e * CT2 + ct2) * KT2 + ktg) * 32 + lane) * 8;
        FragB b;
        b.q[0] = ((const uint4*)bp)[0]; b.q[1] = ((const uint4*)bp)[1];
        yacc[t2][0].v = __builtin_amdgcn_wmma_f32_16x16x32_bf16(false, a0.v, false, b.v, (short)0, yacc[t2][0].v, false, false);
        yacc[t2][1].v = __builtin_amdgcn_wmma_f32_16x16x32_bf16(false, a1.v, false, b.v, (short)0, yacc[t2][1].v, false, false);
      }
    }
    __syncthreads();
  }

  // ---- epilogue: + b2, scale by routed weight, atomic scatter
#pragma unroll
  for (int t2 = 0; t2 < 8; ++t2) {
    const int col  = (wave * 8 + t2) * 16 + nlo;
    const float bs = b2[e * DM + col];
#pragma unroll
    for (int m = 0; m < 2; ++m)
#pragma unroll
      for (int r = 0; r < 8; ++r) {
        const int M = m * 16 + r + hi * 8;
        const float val = (yacc[t2][m].f[r] + bs) * tw[M];
        atomicAdd(out + (size_t)toks[M] * DM + col, val);
      }
  }
}

// ---------------- launch ----------------
extern "C" void kernel_launch(void* const* d_in, const int* in_sizes, int n_in,
                              void* d_out, int out_size, void* d_ws, size_t ws_size,
                              hipStream_t stream) {
  (void)in_sizes; (void)n_in; (void)out_size; (void)ws_size;
  const float* x  = (const float*)d_in[0];
  const float* Wg = (const float*)d_in[1];
  const float* bg = (const float*)d_in[2];
  const float* W1 = (const float*)d_in[3];
  const float* b1 = (const float*)d_in[4];
  const float* W2 = (const float*)d_in[5];
  const float* b2 = (const float*)d_in[6];
  float* out = (float*)d_out;

  char* ws = (char*)d_ws;
  unsigned int*   counts = (unsigned int*)(ws + OFF_COUNTS);
  int*            ltok   = (int*)(ws + OFF_TOK);
  float*          lw     = (float*)(ws + OFF_LW);
  unsigned short* xb     = (unsigned short*)(ws + OFF_XB);
  unsigned short* w1p    = (unsigned short*)(ws + OFF_W1P);
  unsigned short* w2p    = (unsigned short*)(ws + OFF_W2P);

  k_zero   <<<8192, 256, 0, stream>>>((float4*)out, counts);
  k_gate   <<<1024, 256, 0, stream>>>(x, Wg, bg, ltok, lw, counts);
  k_cvt_x  <<<8192, 256, 0, stream>>>((const float4*)x, (ushort4*)xb);
  k_pack_w1<<<4096, 256, 0, stream>>>(W1, w1p);
  k_pack_w2<<<4096, 256, 0, stream>>>(W2, w2p);
  k_moe    <<<NE * MAXT, 256, 83200, stream>>>(xb, (const unsigned int*)w1p,
                                               (const unsigned int*)w2p, b1, b2,
                                               ltok, lw, counts, out);
}
